// EncodeProcessDecode_5317169513193
// MI455X (gfx1250) — matrix operand order repaired and checked by reading, not compiled, not run on hardware
//
#include <hip/hip_runtime.h>

// ---------------------------------------------------------------------------
// CDNA5 (gfx1250) fused EncodeProcessDecode GNN.
// All dense math via v_wmma_f32_16x16x32_f16 (wave32, 16-row tiles per wave).
// Memory-bound workload: f16 operand storage, fused gathers, fixed-point
// int64 atomics for deterministic segment-mean.
// ---------------------------------------------------------------------------

typedef __attribute__((ext_vector_type(16))) _Float16 v16h;
typedef __attribute__((ext_vector_type(8)))  _Float16 v8h;
typedef __attribute__((ext_vector_type(8)))  float    v8f;

#define LDS_FENCE() asm volatile("s_wait_dscnt 0" ::: "memory")

// A fragment (16x32 f16, M x K) from row-major f16 buffer, leading dim ld.
// lane l: row m = l&15, kbase = (l>>4)*8 ; elems 0..7 -> K kb..kb+7, 8..15 -> K 16+kb..
static __device__ __forceinline__ v16h load_a(const _Float16* __restrict__ base, int ld, int kt) {
  int l = threadIdx.x & 31;
  const _Float16* p = base + (l & 15) * ld + kt + ((l >> 4) << 3);
  v8h lo = *(const v8h*)p;
  v8h hi = *(const v8h*)(p + 16);
  v16h r;
#pragma unroll
  for (int i = 0; i < 8; ++i) { r[i] = lo[i]; r[i + 8] = hi[i]; }
  return r;
}

// B fragment (32x16 f16, K x N) from pre-packed weights:
// P layout [kt][nt][lane 0..31][16 f16]  (32 contiguous bytes per lane).
static __device__ __forceinline__ v16h load_b(const _Float16* __restrict__ P, int ntiles, int kt, int nt) {
  int l = threadIdx.x & 31;
  return *(const v16h*)(P + (((size_t)(kt * ntiles + nt) * 32 + l) << 4));
}

static __device__ __forceinline__ v8f wmma16(v16h a, v16h b, v8f c) {
  return __builtin_amdgcn_wmma_f32_16x16x32_f16(false, a, false, b, (short)0, c, false, false);
}

static __device__ __forceinline__ v8f gemm_acc(const _Float16* A, int ldA, int kchunks,
                                               const _Float16* __restrict__ Wp, int ntiles, int nt) {
  v8f acc = {};
  for (int kt = 0; kt < kchunks; ++kt)
    acc = wmma16(load_a(A, ldA, kt * 32), load_b(Wp, ntiles, kt, nt), acc);
  return acc;
}

// H = relu(A @ W + b) written to LDS as f16 (row-major, ld = ldH).
// D layout: lane l col n = nt*16+(l&15); vgpr r row m = (l>>4)*8 + r.
static __device__ __forceinline__ void gemm_relu_h(const _Float16* A, int ldA, int kchunks,
                                                   const _Float16* __restrict__ Wp,
                                                   const float* __restrict__ bias,
                                                   int ntiles, _Float16* H, int ldH) {
  int l = threadIdx.x & 31;
  int mb = (l >> 4) << 3;
  for (int nt = 0; nt < ntiles; ++nt) {
    v8f acc = gemm_acc(A, ldA, kchunks, Wp, ntiles, nt);
    float bs = bias[nt * 16 + (l & 15)];
#pragma unroll
    for (int r = 0; r < 8; ++r) {
      float h = acc[r] + bs;
      H[(mb + r) * ldH + nt * 16 + (l & 15)] = (_Float16)(h > 0.f ? h : 0.f);
    }
  }
}

// --------------------------- utility kernels -------------------------------

__global__ void k_zero32(unsigned int* __restrict__ p, long long n) {
  long long i = (long long)blockIdx.x * blockDim.x + threadIdx.x;
  if (i < n) p[i] = 0u;
}

__global__ void k_count(const int* __restrict__ dst, int* __restrict__ cnt, int E) {
  int e = blockIdx.x * blockDim.x + threadIdx.x;
  if (e < E) atomicAdd(&cnt[dst[e]], 1);
}

__global__ void k_inv(const int* __restrict__ cnt, float* __restrict__ inv, int N) {
  int i = blockIdx.x * blockDim.x + threadIdx.x;
  if (i < N) inv[i] = 1.0f / fmaxf((float)cnt[i], 1.0f);
}

// Repack f32 row-major weight [K][N] -> f16 B-fragment layout, zero padded.
__global__ void k_repack(const float* __restrict__ W, _Float16* __restrict__ P,
                         int K, int N, int Kp, int Np) {
  int idx = blockIdx.x * blockDim.x + threadIdx.x;
  int ntiles = Np >> 4;
  int total = (Kp >> 5) * ntiles * 512;
  if (idx >= total) return;
  int e = idx & 15;
  int lane = (idx >> 4) & 31;
  int f = idx >> 9;
  int nt = f % ntiles, kt = f / ntiles;
  int k = kt * 32 + ((lane >> 4) << 3) + (e & 7) + ((e >> 3) << 4);
  int n = nt * 16 + (lane & 15);
  float v = (k < K && n < N) ? W[(size_t)k * N + n] : 0.0f;
  P[idx] = (_Float16)v;
}

// ------------------------------- encoders ----------------------------------

__global__ void __launch_bounds__(128)
k_encode_node(const float* __restrict__ nodeF, const float* __restrict__ emb,
              const _Float16* __restrict__ W0p, const float* __restrict__ b0,
              const _Float16* __restrict__ W1p, const float* __restrict__ b1,
              _Float16* __restrict__ node_f16, float* __restrict__ node_tail, int N) {
  const int LDA = 72, LDH = 136;
  __shared__ _Float16 sA[4][16 * LDA];
  __shared__ _Float16 sH[4][16 * LDH];
  int w = threadIdx.x >> 5, l = threadIdx.x & 31;
  int tile = blockIdx.x * 4 + w;
  if (tile >= ((N + 15) >> 4)) return;
  int n0 = tile << 4;
  _Float16* A = sA[w]; _Float16* H = sH[w];
  // A[16][64]: cols 0..30 nodeFeatures, 31..46 emb[ptype], 47..63 zero pad.
  for (int c = l; c < 16 * 64; c += 32) {
    int row = c >> 6, col = c & 63;
    int n = n0 + row; if (n >= N) n = N - 1;
    float v = 0.f;
    if (col < 31) v = nodeF[(size_t)n * 32 + col];
    else if (col < 47) {
      int pt = (int)(nodeF[(size_t)n * 32 + 31] + 0.5f);
      v = emb[pt * 16 + (col - 31)];
    }
    A[row * LDA + col] = (_Float16)v;
  }
  LDS_FENCE();
  gemm_relu_h(A, LDA, 2, W0p, b0, 8, H, LDH);   // 47(64) -> 128, relu
  LDS_FENCE();
  int mb = (l >> 4) << 3;
  for (int nt = 0; nt < 8; ++nt) {               // 128 -> 128, linear
    v8f acc = gemm_acc(H, LDH, 4, W1p, 8, nt);
    int col = nt * 16 + (l & 15);
    float bs = b1[col];
#pragma unroll
    for (int r = 0; r < 8; ++r) {
      int n = n0 + mb + r;
      if (n < N) {
        float v = acc[r] + bs;
        node_f16[(size_t)n * 128 + col] = (_Float16)v;
        if (col >= 64) node_tail[(size_t)n * 64 + (col - 64)] = v;
      }
    }
  }
}

__global__ void __launch_bounds__(128)
k_encode_edge(const float* __restrict__ edgeF,
              const _Float16* __restrict__ W0p, const float* __restrict__ b0,
              const _Float16* __restrict__ W1p, const float* __restrict__ b1,
              float* __restrict__ edge_tail, _Float16* __restrict__ edge_tail_h, int E) {
  const int LDA = 40, LDH = 136;
  __shared__ _Float16 sA[4][16 * LDA];
  __shared__ _Float16 sH[4][16 * LDH];
  int w = threadIdx.x >> 5, l = threadIdx.x & 31;
  int tile = blockIdx.x * 4 + w;
  if (tile >= ((E + 15) >> 4)) return;
  int e0 = tile << 4;
  _Float16* A = sA[w]; _Float16* H = sH[w];
  for (int c = l; c < 16 * 32; c += 32) {        // A[16][32]: 3 real cols
    int row = c >> 5, col = c & 31;
    int e = e0 + row; if (e >= E) e = E - 1;
    float v = (col < 3) ? edgeF[(size_t)e * 3 + col] : 0.f;
    A[row * LDA + col] = (_Float16)v;
  }
  LDS_FENCE();
  gemm_relu_h(A, LDA, 1, W0p, b0, 8, H, LDH);    // 3(32) -> 128, relu
  LDS_FENCE();
  int mb = (l >> 4) << 3;
  for (int nt = 4; nt < 8; ++nt) {               // only cols 64..127 ever needed
    v8f acc = gemm_acc(H, LDH, 4, W1p, 8, nt);
    int col = nt * 16 + (l & 15);
    float bs = b1[col];
    int j = col - 64;
#pragma unroll
    for (int r = 0; r < 8; ++r) {
      int e = e0 + mb + r;
      if (e < E) {
        float v = acc[r] + bs;
        edge_tail[(size_t)e * 64 + j] = v;
        edge_tail_h[(size_t)e * 64 + j] = (_Float16)v;
      }
    }
  }
}

// ------------------------------- core step ---------------------------------

__global__ void __launch_bounds__(128)
k_core_edge(const int* __restrict__ srcI, const int* __restrict__ dstI,
            const _Float16* __restrict__ node_f16,
            float* __restrict__ edge_tail, _Float16* __restrict__ edge_tail_h,
            const _Float16* __restrict__ W0p, const float* __restrict__ b0,
            const _Float16* __restrict__ W1p, const float* __restrict__ b1,
            unsigned long long* __restrict__ agg, int E) {
  const int LDA = 200, LDH = 136;
  __shared__ _Float16 sA[4][16 * LDA];
  __shared__ _Float16 sH[4][16 * LDH];
  int w = threadIdx.x >> 5, l = threadIdx.x & 31;
  int tile = blockIdx.x * 4 + w;
  if (tile >= ((E + 15) >> 4)) return;
  int e0 = tile << 4;
  _Float16* A = sA[w]; _Float16* H = sH[w];
  // A[16][192] = [edge_tail | node[dst].tail | node[src].tail], 8-f16 chunks.
  for (int c = l; c < 16 * 24; c += 32) {
    int row = c / 24, ch = c % 24;
    int e = e0 + row; if (e >= E) e = E - 1;
    const _Float16* sp;
    if (ch < 8)       sp = edge_tail_h + (size_t)e * 64 + ch * 8;
    else if (ch < 16) sp = node_f16 + (size_t)dstI[e] * 128 + 64 + (ch - 8) * 8;
    else              sp = node_f16 + (size_t)srcI[e] * 128 + 64 + (ch - 16) * 8;
    *(v8h*)(A + row * LDA + ch * 8) = *(const v8h*)sp;
  }
  LDS_FENCE();
  gemm_relu_h(A, LDA, 6, W0p, b0, 8, H, LDH);    // 192 -> 128, relu
  LDS_FENCE();
  int mb = (l >> 4) << 3;
  for (int nt = 0; nt < 4; ++nt) {               // 128 -> 64 + residual + aggregate
    v8f acc = gemm_acc(H, LDH, 4, W1p, 4, nt);
    int n = nt * 16 + (l & 15);
    float bs = b1[n];
#pragma unroll
    for (int r = 0; r < 8; ++r) {
      int e = e0 + mb + r;
      if (e < E) {
        float v = edge_tail[(size_t)e * 64 + n] + acc[r] + bs;
        edge_tail[(size_t)e * 64 + n] = v;
        edge_tail_h[(size_t)e * 64 + n] = (_Float16)v;
        // deterministic fixed-point segment sum (order-independent int64 atomics)
        long long q = (long long)rintf(v * 1048576.0f);
        atomicAdd(&agg[(size_t)dstI[e] * 64 + n], (unsigned long long)q);
      }
    }
  }
}

__global__ void __launch_bounds__(128)
k_core_node(const unsigned long long* __restrict__ agg, const float* __restrict__ invc,
            _Float16* __restrict__ node_f16, float* __restrict__ node_tail,
            const _Float16* __restrict__ W0p, const float* __restrict__ b0,
            const _Float16* __restrict__ W1p, const float* __restrict__ b1, int N) {
  const int LDA = 136, LDH = 136;
  __shared__ _Float16 sA[4][16 * LDA];
  __shared__ _Float16 sH[4][16 * LDH];
  int w = threadIdx.x >> 5, l = threadIdx.x & 31;
  int tile = blockIdx.x * 4 + w;
  if (tile >= ((N + 15) >> 4)) return;
  int n0 = tile << 4;
  _Float16* A = sA[w]; _Float16* H = sH[w];
  // A[16][128] = [agg_mean(tail) | node.tail]
  for (int c = l; c < 16 * 128; c += 32) {
    int row = c >> 7, col = c & 127;
    int n = n0 + row; if (n >= N) n = N - 1;
    float v;
    if (col < 64)
      v = (float)(long long)agg[(size_t)n * 64 + col] * (1.0f / 1048576.0f) * invc[n];
    else
      v = (float)node_f16[(size_t)n * 128 + col];
    A[row * LDA + col] = (_Float16)v;
  }
  LDS_FENCE();
  gemm_relu_h(A, LDA, 4, W0p, b0, 8, H, LDH);    // 128 -> 128, relu
  LDS_FENCE();
  int mb = (l >> 4) << 3;
  for (int nt = 0; nt < 4; ++nt) {               // 128 -> 64 + residual
    v8f acc = gemm_acc(H, LDH, 4, W1p, 4, nt);
    int col = nt * 16 + (l & 15);
    float bs = b1[col];
#pragma unroll
    for (int r = 0; r < 8; ++r) {
      int n = n0 + mb + r;
      if (n < N) {
        float v = node_tail[(size_t)n * 64 + col] + acc[r] + bs;
        node_tail[(size_t)n * 64 + col] = v;
        node_f16[(size_t)n * 128 + 64 + col] = (_Float16)v;
      }
    }
  }
}

// -------------------------------- decoder ----------------------------------

__global__ void __launch_bounds__(128)
k_decode(const _Float16* __restrict__ node_f16,
         const _Float16* __restrict__ W0p, const float* __restrict__ b0,
         const _Float16* __restrict__ W1p, const float* __restrict__ b1,
         float* __restrict__ out, int N) {
  const int LDA = 136, LDH = 136;
  __shared__ _Float16 sA[4][16 * LDA];
  __shared__ _Float16 sH[4][16 * LDH];
  int w = threadIdx.x >> 5, l = threadIdx.x & 31;
  int tile = blockIdx.x * 4 + w;
  if (tile >= ((N + 15) >> 4)) return;
  int n0 = tile << 4;
  _Float16* A = sA[w]; _Float16* H = sH[w];
  for (int c = l; c < 16 * 16; c += 32) {        // stage full node row (16 x 128)
    int row = c >> 4, ch = c & 15;
    int n = n0 + row; if (n >= N) n = N - 1;
    *(v8h*)(A + row * LDA + ch * 8) = *(const v8h*)(node_f16 + (size_t)n * 128 + ch * 8);
  }
  LDS_FENCE();
  gemm_relu_h(A, LDA, 4, W0p, b0, 8, H, LDH);    // 128 -> 128, relu
  LDS_FENCE();
  int mb = (l >> 4) << 3;
  v8f acc = gemm_acc(H, LDH, 4, W1p, 1, 0);      // 128 -> 3 (padded to 16)
  int col = l & 15;
  float bs = (col < 3) ? b1[col] : 0.f;
#pragma unroll
  for (int r = 0; r < 8; ++r) {
    int n = n0 + mb + r;
    if (n < N && col < 3) out[(size_t)n * 3 + col] = acc[r] + bs;
  }
}

// ------------------------------- launcher ----------------------------------

extern "C" void kernel_launch(void* const* d_in, const int* in_sizes, int n_in,
                              void* d_out, int out_size, void* d_ws, size_t ws_size,
                              hipStream_t stream) {
  (void)n_in; (void)out_size; (void)ws_size;
  const int E = in_sizes[0] / 2;
  const int N = in_sizes[1] / 32;

  const int*   ei    = (const int*)d_in[0];
  const int*   srcI  = ei;          // edge_index[0] = j (source)
  const int*   dstI  = ei + E;      // edge_index[1] = i (target)
  const float* nodeF = (const float*)d_in[1];
  const float* edgeF = (const float*)d_in[2];
  const float* emb   = (const float*)d_in[3];

  const float *enW0 = (const float*)d_in[4],  *enB0 = (const float*)d_in[5];
  const float *enW1 = (const float*)d_in[6],  *enB1 = (const float*)d_in[7];
  const float *eeW0 = (const float*)d_in[8],  *eeB0 = (const float*)d_in[9];
  const float *eeW1 = (const float*)d_in[10], *eeB1 = (const float*)d_in[11];
  const float *ceW0 = (const float*)d_in[12], *ceB0 = (const float*)d_in[13];
  const float *ceW1 = (const float*)d_in[14], *ceB1 = (const float*)d_in[15];
  const float *cnW0 = (const float*)d_in[16], *cnB0 = (const float*)d_in[17];
  const float *cnW1 = (const float*)d_in[18], *cnB1 = (const float*)d_in[19];
  const float *dW0  = (const float*)d_in[20], *dB0  = (const float*)d_in[21];
  const float *dW1  = (const float*)d_in[22], *dB1  = (const float*)d_in[23];

  // ---- workspace carve-out (256B aligned) ----
  char* ws = (char*)d_ws;
  size_t off = 0;
  auto take = [&](size_t bytes) { size_t o = off; off += (bytes + 255) & ~(size_t)255; return o; };
  _Float16* node_f16     = (_Float16*)(ws + take((size_t)N * 128 * 2));
  float*    node_tail    = (float*)   (ws + take((size_t)N * 64 * 4));
  float*    edge_tail    = (float*)   (ws + take((size_t)E * 64 * 4));
  _Float16* edge_tail_h  = (_Float16*)(ws + take((size_t)E * 64 * 2));
  unsigned long long* agg = (unsigned long long*)(ws + take((size_t)N * 64 * 8));
  int*      cnt          = (int*)     (ws + take((size_t)N * 4));
  float*    invc         = (float*)   (ws + take((size_t)N * 4));
  _Float16* P_enW0 = (_Float16*)(ws + take((size_t)64  * 128 * 2));
  _Float16* P_enW1 = (_Float16*)(ws + take((size_t)128 * 128 * 2));
  _Float16* P_eeW0 = (_Float16*)(ws + take((size_t)32  * 128 * 2));
  _Float16* P_eeW1 = (_Float16*)(ws + take((size_t)128 * 128 * 2));
  _Float16* P_ceW0 = (_Float16*)(ws + take((size_t)192 * 128 * 2));
  _Float16* P_ceW1 = (_Float16*)(ws + take((size_t)128 * 64 * 2));
  _Float16* P_cnW0 = (_Float16*)(ws + take((size_t)128 * 128 * 2));
  _Float16* P_cnW1 = (_Float16*)(ws + take((size_t)128 * 64 * 2));
  _Float16* P_dW0  = (_Float16*)(ws + take((size_t)128 * 128 * 2));
  _Float16* P_dW1  = (_Float16*)(ws + take((size_t)128 * 16 * 2));

  auto repack = [&](const float* W, _Float16* P, int K, int Nn, int Kp, int Np) {
    int total = (Kp >> 5) * (Np >> 4) * 512;
    k_repack<<<(total + 255) / 256, 256, 0, stream>>>(W, P, K, Nn, Kp, Np);
  };
  repack(enW0, P_enW0, 47, 128, 64, 128);
  repack(enW1, P_enW1, 128, 128, 128, 128);
  repack(eeW0, P_eeW0, 3, 128, 32, 128);
  repack(eeW1, P_eeW1, 128, 128, 128, 128);
  repack(ceW0, P_ceW0, 192, 128, 192, 128);
  repack(ceW1, P_ceW1, 128, 64, 128, 64);
  repack(cnW0, P_cnW0, 128, 128, 128, 128);
  repack(cnW1, P_cnW1, 128, 64, 128, 64);
  repack(dW0,  P_dW0,  128, 128, 128, 128);
  repack(dW1,  P_dW1,  128, 3, 128, 16);

  // degree counts (deterministic integer atomics)
  k_zero32<<<(N + 255) / 256, 256, 0, stream>>>((unsigned int*)cnt, N);
  k_count<<<(E + 255) / 256, 256, 0, stream>>>(dstI, cnt, E);
  k_inv<<<(N + 255) / 256, 256, 0, stream>>>(cnt, invc, N);

  const int nTilesN = (N + 15) >> 4, nTilesE = (E + 15) >> 4;
  const int blkN = (nTilesN + 3) / 4, blkE = (nTilesE + 3) / 4;

  k_encode_node<<<blkN, 128, 0, stream>>>(nodeF, emb, P_enW0, enB0, P_enW1, enB1,
                                          node_f16, node_tail, N);
  k_encode_edge<<<blkE, 128, 0, stream>>>(edgeF, P_eeW0, eeB0, P_eeW1, eeB1,
                                          edge_tail, edge_tail_h, E);

  for (int step = 0; step < 3; ++step) {
    long long aggWords = (long long)N * 128;   // N*64 int64 as u32 words
    k_zero32<<<(int)((aggWords + 255) / 256), 256, 0, stream>>>((unsigned int*)agg, aggWords);
    k_core_edge<<<blkE, 128, 0, stream>>>(srcI, dstI, node_f16, edge_tail, edge_tail_h,
                                          P_ceW0, ceB0, P_ceW1, ceB1, agg, E);
    k_core_node<<<blkN, 128, 0, stream>>>(agg, invc, node_f16, node_tail,
                                          P_cnW0, cnB0, P_cnW1, cnB1, N);
  }

  k_decode<<<blkN, 128, 0, stream>>>(node_f16, P_dW0, dB0, P_dW1, dB1, (float*)d_out, N);
}
